// ESN_80367428042799
// MI455X (gfx1250) — matrix operand order repaired and checked by reading, compiled, not run on hardware
//
#include <hip/hip_runtime.h>
#include <hip/hip_bf16.h>
#include <stdint.h>

// Problem constants (from reference setup_inputs)
#define B_   32
#define T_   1024
#define D_   128
#define H_   1024
#define LEAK 0.9f

typedef __attribute__((ext_vector_type(16))) __bf16   v16bf;
typedef __attribute__((ext_vector_type(8)))  __bf16   v8bf;
typedef __attribute__((ext_vector_type(8)))  float    v8f;
typedef __attribute__((ext_vector_type(4)))  unsigned v4u;
typedef __attribute__((ext_vector_type(8)))  int      v8i;
typedef __attribute__((ext_vector_type(4)))  int      v4i;

__device__ __forceinline__ unsigned short f2bf(float f) {
  __bf16 b = (__bf16)f;                       // hardware cvt, RNE
  return __builtin_bit_cast(unsigned short, b);
}

// ---------------------------------------------------------------------------
// Phase 0: zero barrier counter + zero both bf16 h state buffers (128 KB)
// ---------------------------------------------------------------------------
__global__ void k_init(unsigned* __restrict__ counter, unsigned* __restrict__ hbuf, int n) {
  int i = blockIdx.x * blockDim.x + threadIdx.x;
  if (i == 0) *counter = 0u;
  if (i < n) hbuf[i] = 0u;
}

// ---------------------------------------------------------------------------
// Phase 0.5: bulk f32 -> bf16 convert (8 elements/thread, packed hw cvt).
// Used once for x, W_in and W so the GEMM kernels do zero conversion VALU.
// ---------------------------------------------------------------------------
__global__ void k_cvt(const float* __restrict__ src, unsigned short* __restrict__ dst) {
  const size_t i = ((size_t)blockIdx.x * blockDim.x + threadIdx.x) * 8;
  v8f  f = *(const v8f*)(src + i);
  v8bf b = __builtin_convertvector(f, v8bf);
  *(v8bf*)(dst + i) = b;
}

// ---------------------------------------------------------------------------
// Phase 1: pre_in = x @ W_in^T + bias  -> written straight into d_out (B,T,H).
// One 16x16 tile per wave, K = 128 (4 x wmma_f32_16x16x32_bf16), bf16 inputs.
// ---------------------------------------------------------------------------
__global__ void __launch_bounds__(256) k_proj(const unsigned short* __restrict__ xb,
                                              const unsigned short* __restrict__ Winb,
                                              const float* __restrict__ bias,
                                              float* __restrict__ out) {
  const int lane = threadIdx.x & 31;
  const int wave = threadIdx.x >> 5;
  const int tile = blockIdx.x * 8 + wave;    // (BT/16)*(H/16) = 2048*64 tiles
  const int mt   = tile >> 6;                // 0..2047 : 16-row slab of BT
  const int nt   = tile & 63;                // 0..63   : 16-col slab of H
  const int half = lane >> 4;
  const int lm   = lane & 15;

  const unsigned short* arow = xb   + (size_t)(mt * 16 + lm) * D_;   // A: M = lane%16
  const unsigned short* brow = Winb + (size_t)(nt * 16 + lm) * D_;   // B: N = lane%16

  v8f c = {};
  #pragma unroll
  for (int k0 = 0; k0 < D_; k0 += 32) {
    union { uint4 q[2]; v16bf v; } a, b;
    a.q[0] = *(const uint4*)(arow + k0 + half * 8);          // k = k0+half*8+e
    a.q[1] = *(const uint4*)(arow + k0 + 16 + half * 8);     // k = k0+16+half*8+e
    b.q[0] = *(const uint4*)(brow + k0 + half * 8);          // k = k0+half*8+e
    b.q[1] = *(const uint4*)(brow + k0 + 16 + half * 8);
    c = __builtin_amdgcn_wmma_f32_16x16x32_bf16(false, a.v, false, b.v,
                                                (short)0, c, false, false);
  }
  const int ncol = nt * 16 + lm;             // D-layout: N = lane%16
  const float bs = bias[ncol];
  #pragma unroll
  for (int v = 0; v < 8; ++v) {
    int m = mt * 16 + half * 8 + v;          // D-layout: M = 8*(lane>=16)+v
    out[(size_t)m * H_ + ncol] = c[v] + bs;
  }
}

// ---------------------------------------------------------------------------
// Phase 2: persistent scan kernel. 32 WGs x 256 threads. WG g owns output
// columns [32g, 32g+32). LDS holds its bf16 W slab (32 x 1024 = 64 KB, loaded
// once via TDM) + 4 KB K-split partials. 8 waves = 4 tiles x 2 K-halves.
// Per step: z = h @ W^T (WMMA), h = 0.1*h + 0.9*tanh(pre_in + z), grid sync.
// ---------------------------------------------------------------------------
__global__ void __launch_bounds__(256) k_scan(float* out,                    // aliases pre_in
                                              const unsigned short* __restrict__ Wb,
                                              unsigned short* h0,
                                              unsigned short* h1,
                                              unsigned* counter, int nblk) {
  extern __shared__ unsigned char smem[];
  unsigned short* ldsB = (unsigned short*)smem;                 // [32][1024] bf16 = 64 KB
  float*          ldsP = (float*)(smem + 32 * 1024 * 2);        // [4][32][8]  f32  = 4 KB

  const int lane  = threadIdx.x & 31;
  const int wave  = threadIdx.x >> 5;
  const int tile  = wave >> 1;           // 0..3
  const int khalf = wave & 1;            // K split: [0,512) / [512,1024)
  const int mtile = tile & 1;            // batch rows 0-15 / 16-31
  const int ntile = tile >> 1;           // 16-col tile within 32-col slab
  const int half  = lane >> 4;
  const int lm    = lane & 15;
  const int nbase = blockIdx.x * 32 + ntile * 16;

  // ---- load 64 KB contiguous W slab (rows 32g..32g+31 of Wb) into LDS ----
  const unsigned short* gB = Wb + (size_t)(blockIdx.x * 32) * H_;
#if __has_builtin(__builtin_amdgcn_tensor_load_to_lds) && __has_builtin(__builtin_amdgcn_s_wait_tensorcnt)
  if (wave == 0) {
    unsigned long long ga = (unsigned long long)(uintptr_t)gB;
    unsigned ldsOff = (unsigned)(uintptr_t)ldsB;
    // D# group0: count=1 | lds_addr | global_addr[56:0] | type=2
    v4u g0;
    g0.x = 1u;
    g0.y = ldsOff;
    g0.z = (unsigned)(ga & 0xFFFFFFFFull);
    g0.w = (unsigned)((ga >> 32) & 0x1FFFFFFull) | (2u << 30);
    // D# group1: data_size=3 (8B), 1-D: tensor_dim0 = tile_dim0 = 8192 elems (64 KB)
    v8i g1 = {0, 0, 0, 0, 0, 0, 0, 0};
    g1[0] = (int)(3u << 16);            // data_size = 8 bytes
    g1[1] = (int)(8192u << 16);         // tensor_dim0[15:0] (bits [63:48])
    g1[3] = (int)(8192u << 16);         // tile_dim0 (bits [127:112])
    v4i z4 = {0, 0, 0, 0};
    v8i z8 = {0, 0, 0, 0, 0, 0, 0, 0};
    __builtin_amdgcn_tensor_load_to_lds(g0, g1, z4, z4, z8, 0);
    __builtin_amdgcn_s_wait_tensorcnt(0);
  }
  __syncthreads();
#else
  {
    const uint4* src = (const uint4*)gB;
    uint4*       dst = (uint4*)ldsB;
    for (int i = threadIdx.x; i < (32 * 1024 * 2) / 16; i += 256) dst[i] = src[i];
  }
  __syncthreads();
#endif

  const int arowIdx = mtile * 16 + lm;                       // batch row (A operand, M=lane%16)
  const unsigned short* brow = ldsB + (size_t)(ntile * 16 + lm) * H_;  // B col lm's k-vector
  const int kbeg = khalf * 512;

  float hstate[8];
  #pragma unroll
  for (int v = 0; v < 8; ++v) hstate[v] = 0.0f;

  for (int t = 0; t < T_; ++t) {
    const unsigned short* hcur = (t & 1) ? h1 : h0;
    unsigned short*       hnxt = (t & 1) ? h0 : h1;
    const unsigned short* arow = hcur + (size_t)arowIdx * H_;

    v8f c = {};
    #pragma unroll 4
    for (int kk = 0; kk < 512; kk += 32) {
      const int k0 = kbeg + kk;
      union { uint4 q[2]; v16bf v; } a, b;
      a.q[0] = *(const uint4*)(arow + k0 + half * 8);          // k = k0+half*8+e
      a.q[1] = *(const uint4*)(arow + k0 + 16 + half * 8);     // k = k0+16+half*8+e
      b.q[0] = *(const uint4*)(brow + k0 + half * 8);          // k = k0+half*8+e
      b.q[1] = *(const uint4*)(brow + k0 + 16 + half * 8);
      c = __builtin_amdgcn_wmma_f32_16x16x32_bf16(false, a.v, false, b.v,
                                                  (short)0, c, false, false);
    }

    // K-split reduction: khalf==1 parks its partial in LDS, khalf==0 reduces.
    if (khalf == 1) {
      float* p = ldsP + ((size_t)tile * 32 + lane) * 8;
      #pragma unroll
      for (int v = 0; v < 8; ++v) p[v] = c[v];
    }
    __syncthreads();
    if (khalf == 0) {
      const float* p = ldsP + ((size_t)tile * 32 + lane) * 8;
      const int ncol = nbase + lm;
      #pragma unroll
      for (int v = 0; v < 8; ++v) {
        const int m = mtile * 16 + half * 8 + v;               // batch index
        const size_t oidx = ((size_t)m * T_ + t) * H_ + ncol;
        const float z    = c[v] + p[v];
        const float pre  = out[oidx];                          // pre_in (in-place)
        const float hn   = tanhf(pre + z);
        const float hnew = (1.0f - LEAK) * hstate[v] + LEAK * hn;
        hstate[v] = hnew;
        out[oidx] = hnew;                                      // state stream -> output
        hnxt[(size_t)m * H_ + ncol] = f2bf(hnew);              // next-step A operand
      }
    }

    // Grid-wide barrier (monotonic counter; all 32 WGs resident/persistent)
    __syncthreads();
    if (threadIdx.x == 0) {
      __threadfence();
      atomicAdd(counter, 1u);
      const unsigned tgt = (unsigned)(t + 1) * (unsigned)nblk;
      while (__hip_atomic_load(counter, __ATOMIC_ACQUIRE, __HIP_MEMORY_SCOPE_AGENT) < tgt)
        __builtin_amdgcn_s_sleep(2);
    }
    __syncthreads();
  }
}

// ---------------------------------------------------------------------------
// ws layout (bytes):
//   [0,256)                  barrier counter
//   [256, +64K)              h ping (bf16)
//   [+64K, +128K)            h pong (bf16)
//   [+128K, +2.125M)         W  bf16 (2 MB)
//   [+2.125M, +10.125M)      x  bf16 (8 MB)
//   [+10.125M, +10.375M)     W_in bf16 (256 KB)        total ~10.4 MB
// ---------------------------------------------------------------------------
extern "C" void kernel_launch(void* const* d_in, const int* in_sizes, int n_in,
                              void* d_out, int out_size, void* d_ws, size_t ws_size,
                              hipStream_t stream) {
  const float* x    = (const float*)d_in[0];   // (32,1024,128)
  const float* Win  = (const float*)d_in[1];   // (1024,128)
  const float* W    = (const float*)d_in[2];   // (1024,1024)
  const float* bias = (const float*)d_in[3];   // (1024,)
  float* out = (float*)d_out;                  // (32,1024,1024)

  unsigned char* ws = (unsigned char*)d_ws;
  unsigned*       counter = (unsigned*)ws;
  unsigned short* h0   = (unsigned short*)(ws + 256);
  unsigned short* h1   = (unsigned short*)(ws + 256 + 65536);
  unsigned short* Wb   = (unsigned short*)(ws + 256 + 131072);
  unsigned short* xb   = (unsigned short*)(ws + 256 + 131072 + 2097152);
  unsigned short* Winb = (unsigned short*)(ws + 256 + 131072 + 2097152 + 8388608);

  k_init<<<128, 256, 0, stream>>>(counter, (unsigned*)h0, (2 * B_ * H_ * 2) / 4);

  k_cvt<<<(H_ * H_)      / (8 * 256), 256, 0, stream>>>(W,   Wb);    // 512 blocks
  k_cvt<<<(B_ * T_ * D_) / (8 * 256), 256, 0, stream>>>(x,   xb);    // 2048 blocks
  k_cvt<<<(H_ * D_)      / (8 * 256), 256, 0, stream>>>(Win, Winb);  // 64 blocks

  k_proj<<<(B_ * T_ / 16) * (H_ / 16) / 8, 256, 0, stream>>>(xb, Winb, bias, out);

  const int nblk = H_ / 32;                                  // 32 persistent WGs
  const size_t smem = 32 * 1024 * 2 + 4 * 32 * 8 * 4;        // 64 KB W slab + 4 KB partials
  k_scan<<<nblk, 256, smem, stream>>>(out, Wb, h0, h1, counter, nblk);
}
